// SelfAttention_87548613362098
// MI455X (gfx1250) — matrix operand order repaired and checked by reading, not compiled, not run on hardware
//
#include <hip/hip_runtime.h>
#include <hip/hip_bf16.h>

// ---------------------------------------------------------------------------
// Pre-norm self-attention block for MI455X (gfx1250), bf16 WMMA pipeline.
//   h = LN(x)             -> bf16
//   q,k,v = h @ W + b     -> bf16  (WMMA GEMM, async-to-LDS double buffered)
//   o = flash-attn(q,k,v) -> bf16  (WMMA QK^T / PV, DPP online softmax)
//   out = x + o @ Wo + bo -> f32   (WMMA GEMM with residual)
// ---------------------------------------------------------------------------

#define HID   1024
#define NH    16
#define HD    64
#define BATCH 4
#define SEQ   2048
#define MROWS (BATCH * SEQ)   // 8192
#define ATT_SCALE 0.03125f    // 1024^-0.5 == 1/32 (exact in bf16)

typedef __bf16 bf16_t;
typedef __attribute__((ext_vector_type(16))) __bf16 v16bf;
typedef __attribute__((ext_vector_type(8)))  __bf16 v8bf;
typedef __attribute__((ext_vector_type(8)))  float  v8f;

// --- CDNA5 async memory->LDS (GVS mode: SGPR64 base + VGPR i32 offset) -----
#define ASYNC_LOAD_B128(ldsoff, base, voff)                                   \
  asm volatile("global_load_async_to_lds_b128 %0, %1, %2"                     \
               :: "v"(ldsoff), "v"(voff), "s"(base) : "memory")
#define WAIT_ASYNC(n) asm volatile("s_wait_asynccnt " #n ::: "memory")

__device__ __forceinline__ unsigned lds_off(const void* p) {
  return (unsigned)(size_t)p;   // LDS aperture: low 32 bits = LDS byte offset
}

__device__ __forceinline__ bf16_t f2bf(float f) {
  unsigned u = __builtin_bit_cast(unsigned, f);
  unsigned r = u + 0x7FFFu + ((u >> 16) & 1u);   // round-to-nearest-even
  unsigned short hs = (unsigned short)(r >> 16);
  return __builtin_bit_cast(bf16_t, hs);
}
__device__ __forceinline__ float bf2f(bf16_t b) {
  unsigned short hs = __builtin_bit_cast(unsigned short, b);
  unsigned u = ((unsigned)hs) << 16;
  return __builtin_bit_cast(float, u);
}

__device__ __forceinline__ v16bf cat8(v8bf lo, v8bf hi) {
  v16bf r;
#pragma unroll
  for (int i = 0; i < 8; ++i) { r[i] = lo[i]; r[i + 8] = hi[i]; }
  return r;
}
__device__ __forceinline__ v16bf ldfrag2(const bf16_t* p0, const bf16_t* p1) {
  return cat8(*(const v8bf*)p0, *(const v8bf*)p1);
}
__device__ __forceinline__ v16bf ldfrag_seq(const bf16_t* p) {
  return cat8(*(const v8bf*)p, *(const v8bf*)(p + 8));
}

__device__ __forceinline__ v8f wmma_bf16(v16bf a, v16bf b, v8f c) {
  return __builtin_amdgcn_wmma_f32_16x16x32_bf16(
      false, a, false, b, (short)0, c, false, false);
}

// --- 16-lane row reductions via DPP16 (VALU, no LDS traffic) ---------------
// xor1 = quad_perm(1,0,3,2)=0xB1, xor2 = quad_perm(2,3,0,1)=0x4E,
// then row_half_mirror(0x141) and row_mirror(0x140) once quads are uniform.
template <int CTRL>
__device__ __forceinline__ float dpp_step(float x) {
  int xi = __builtin_amdgcn_update_dpp(0, __builtin_bit_cast(int, x),
                                       CTRL, 0xF, 0xF, true);
  return __builtin_bit_cast(float, xi);
}
__device__ __forceinline__ float row_max16(float x) {
  x = fmaxf(x, dpp_step<0xB1>(x));
  x = fmaxf(x, dpp_step<0x4E>(x));
  x = fmaxf(x, dpp_step<0x141>(x));
  x = fmaxf(x, dpp_step<0x140>(x));
  return x;
}
__device__ __forceinline__ float row_sum16(float x) {
  x += dpp_step<0xB1>(x);
  x += dpp_step<0x4E>(x);
  x += dpp_step<0x141>(x);
  x += dpp_step<0x140>(x);
  return x;
}

// ---------------------------------------------------------------------------
// LayerNorm: one block per row, 256 threads, 4 elements/thread -> bf16 out
// ---------------------------------------------------------------------------
__global__ __launch_bounds__(256) void ln_kernel(const float* __restrict__ x,
                                                 const float* __restrict__ gamma,
                                                 const float* __restrict__ beta,
                                                 bf16_t* __restrict__ h) {
  __shared__ float red[256];
  const int row = blockIdx.x;
  const int t = threadIdx.x;
  const float* xr = x + (size_t)row * HID;
  float4 xv = *(const float4*)(xr + t * 4);

  float s = xv.x + xv.y + xv.z + xv.w;
  red[t] = s;
  __syncthreads();
#pragma unroll
  for (int st = 128; st > 0; st >>= 1) {
    if (t < st) red[t] += red[t + st];
    __syncthreads();
  }
  const float mu = red[0] * (1.0f / HID);
  __syncthreads();

  float dx = xv.x - mu, dy = xv.y - mu, dz = xv.z - mu, dw = xv.w - mu;
  red[t] = dx * dx + dy * dy + dz * dz + dw * dw;
  __syncthreads();
#pragma unroll
  for (int st = 128; st > 0; st >>= 1) {
    if (t < st) red[t] += red[t + st];
    __syncthreads();
  }
  const float rstd = rsqrtf(red[0] * (1.0f / HID) + 1e-5f);

  bf16_t* hr = h + (size_t)row * HID + t * 4;
  const float* g4 = gamma + t * 4;
  const float* b4 = beta + t * 4;
  hr[0] = f2bf(dx * rstd * g4[0] + b4[0]);
  hr[1] = f2bf(dy * rstd * g4[1] + b4[1]);
  hr[2] = f2bf(dz * rstd * g4[2] + b4[2]);
  hr[3] = f2bf(dw * rstd * g4[3] + b4[3]);
}

// ---------------------------------------------------------------------------
// W[k][n] f32 -> Wt[n][k] bf16 (tiled transpose via LDS)
// ---------------------------------------------------------------------------
__global__ __launch_bounds__(256) void wtrans_kernel(const float* __restrict__ W,
                                                     bf16_t* __restrict__ Wt) {
  __shared__ float tile[32][33];
  const int n0 = blockIdx.x * 32, k0 = blockIdx.y * 32;
  const int tx = threadIdx.x, ty = threadIdx.y;   // 32 x 8
#pragma unroll
  for (int j = 0; j < 4; ++j)
    tile[ty + j * 8][tx] = W[(size_t)(k0 + ty + j * 8) * HID + n0 + tx];
  __syncthreads();
#pragma unroll
  for (int j = 0; j < 4; ++j)
    Wt[(size_t)(n0 + ty + j * 8) * HID + k0 + tx] = f2bf(tile[tx][ty + j * 8]);
}

// ---------------------------------------------------------------------------
// GEMM: C[M,N] = A[M,K](bf16) x Wt[N,K](bf16) + bias; M=8192, N=K=1024.
// 128x128 tile, BK=32, double-buffered GLOBAL_LOAD_ASYNC_TO_LDS_B128.
// 8 waves in 4x2 grid; each wave: 2x4 WMMA 16x16x32 tiles.
// ---------------------------------------------------------------------------
__device__ __forceinline__ void gemm_stage(bf16_t (*As)[40], bf16_t (*Bs)[40],
                                           const bf16_t* A, const bf16_t* Bt,
                                           int m0, int n0, int k0,
                                           int lr, int seg) {
  const int c = seg * 8;
  ASYNC_LOAD_B128(lds_off(&As[lr][c]),      A,  (unsigned)(((m0 + lr)      * HID + k0 + c) * 2));
  ASYNC_LOAD_B128(lds_off(&As[lr + 64][c]), A,  (unsigned)(((m0 + lr + 64) * HID + k0 + c) * 2));
  ASYNC_LOAD_B128(lds_off(&Bs[lr][c]),      Bt, (unsigned)(((n0 + lr)      * HID + k0 + c) * 2));
  ASYNC_LOAD_B128(lds_off(&Bs[lr + 64][c]), Bt, (unsigned)(((n0 + lr + 64) * HID + k0 + c) * 2));
}

template <bool FINAL>
__global__ __launch_bounds__(256) void gemm128_kernel(
    const bf16_t* __restrict__ A, const bf16_t* __restrict__ Bt,
    const float* __restrict__ bias, const float* __restrict__ resid,
    bf16_t* __restrict__ outb, float* __restrict__ outf) {
  __shared__ __align__(16) bf16_t As[2][128][40];
  __shared__ __align__(16) bf16_t Bs[2][128][40];

  const int m0 = blockIdx.y * 128, n0 = blockIdx.x * 128;
  const int t = threadIdx.x;
  const int lane = t & 31, wave = t >> 5;
  const int wm = wave >> 1, wn = wave & 1;
  const int li = lane & 15, g = lane >> 4;
  const int lr = t >> 2, seg = t & 3;   // 64 rows x 4 16B-segments per pass

  v8f acc[2][4];
#pragma unroll
  for (int i = 0; i < 2; ++i)
#pragma unroll
    for (int j = 0; j < 4; ++j) acc[i][j] = (v8f)0.0f;

  gemm_stage(As[0], Bs[0], A, Bt, m0, n0, 0, lr, seg);

  const int NIT = HID / 32;
  for (int it = 0; it < NIT; ++it) {
    const int cur = it & 1;
    if (it + 1 < NIT) {
      gemm_stage(As[cur ^ 1], Bs[cur ^ 1], A, Bt, m0, n0, (it + 1) * 32, lr, seg);
      WAIT_ASYNC(0x4);   // current tile's 4 loads landed; next 4 in flight
    } else {
      WAIT_ASYNC(0x0);
    }
    __syncthreads();

    v16bf af[2], bfm[4];
#pragma unroll
    for (int mt = 0; mt < 2; ++mt) {
      const int row = wm * 32 + mt * 16 + li;
      af[mt] = ldfrag2(&As[cur][row][8 * g], &As[cur][row][16 + 8 * g]);
    }
#pragma unroll
    for (int nt = 0; nt < 4; ++nt) {
      const int col = wn * 64 + nt * 16 + li;
      bfm[nt] = ldfrag2(&Bs[cur][col][8 * g], &Bs[cur][col][16 + 8 * g]);
    }
#pragma unroll
    for (int mt = 0; mt < 2; ++mt)
#pragma unroll
      for (int nt = 0; nt < 4; ++nt)
        acc[mt][nt] = wmma_bf16(af[mt], bfm[nt], acc[mt][nt]);
    __syncthreads();   // everyone done reading 'cur' before it is re-staged
  }

  // epilogue (C layout: n = lane&15, m = r + 8*(lane>>4))
#pragma unroll
  for (int mt = 0; mt < 2; ++mt)
#pragma unroll
    for (int nt = 0; nt < 4; ++nt)
#pragma unroll
      for (int r = 0; r < 8; ++r) {
        const int m = m0 + wm * 32 + mt * 16 + r + 8 * g;
        const int n = n0 + wn * 64 + nt * 16 + li;
        float v = acc[mt][nt][r] + bias[n];
        if (FINAL)
          outf[(size_t)m * HID + n] = v + resid[(size_t)m * HID + n];
        else
          outb[(size_t)m * HID + n] = f2bf(v);
      }
}

// ---------------------------------------------------------------------------
// Flash attention: grid (SEQ/128, BATCH*NH). 8 waves x 16 q-rows; KV blocks
// of 32; K tile staged via async-to-LDS; DPP online softmax; WMMA everywhere.
// ---------------------------------------------------------------------------
__global__ __launch_bounds__(256) void flash_kernel(const bf16_t* __restrict__ q,
                                                    const bf16_t* __restrict__ k,
                                                    const bf16_t* __restrict__ v,
                                                    bf16_t* __restrict__ o) {
  __shared__ __align__(16) bf16_t Ks[32][64];       // [kv][d]
  __shared__ __align__(16) bf16_t Vs[64][40];       // [d][kv] (transposed)
  __shared__ __align__(16) bf16_t Ps[8][16][40];    // per-wave P round-trip

  const int bh = blockIdx.y;
  const int b = bh / NH, hh = bh % NH;
  const int q0 = blockIdx.x * 128;
  const int t = threadIdx.x, lane = t & 31, wave = t >> 5;
  const int li = lane & 15, g = lane >> 4;
  const int rowbase = b * SEQ;
  const int qr0 = q0 + wave * 16;

  // Q fragments (two d-steps over HD=64), pre-scaled by 1/32 (exact)
  v16bf qf[2];
#pragma unroll
  for (int ks = 0; ks < 2; ++ks) {
    const bf16_t* qp = q + (size_t)(rowbase + qr0 + li) * HID + hh * HD + ks * 32;
    v16bf raw = ldfrag2(qp + 8 * g, qp + 16 + 8 * g);
#pragma unroll
    for (int i = 0; i < 16; ++i) raw[i] = f2bf(bf2f(raw[i]) * ATT_SCALE);
    qf[ks] = raw;
  }

  v8f oacc[4];
#pragma unroll
  for (int i = 0; i < 4; ++i) oacc[i] = (v8f)0.0f;
  float mrow[8], lrow[8];
#pragma unroll
  for (int r = 0; r < 8; ++r) { mrow[r] = -3.0e38f; lrow[r] = 0.0f; }

  const int kv_end = q0 + 128;            // causal upper bound for this block
  const int kr = t >> 3, kseg = t & 7;    // K staging: 32 rows x 8 x 16B
  const int vr = t & 31, vseg = t >> 5;   // V staging: 32 rows x 8 segs

  for (int kv0 = 0; kv0 < kv_end; kv0 += 32) {
    __syncthreads();   // previous tile fully consumed
    // K block [32][64]: one async B128 per thread, memory -> LDS directly
    ASYNC_LOAD_B128(lds_off(&Ks[kr][kseg * 8]), k,
                    (unsigned)(((rowbase + kv0 + kr) * HID + hh * HD + kseg * 8) * 2));
    // V block transposed -> Vs[d][kv] (element scatter, through VGPRs)
    {
      v8bf tv = *(const v8bf*)(v + (size_t)(rowbase + kv0 + vr) * HID + hh * HD + vseg * 8);
#pragma unroll
      for (int i = 0; i < 8; ++i) Vs[vseg * 8 + i][vr] = tv[i];
    }
    WAIT_ASYNC(0x0);
    __syncthreads();

    // S = Q * K^T (two 16-wide kv tiles, two d-steps each)
    v8f s0 = (v8f)0.0f, s1 = (v8f)0.0f;
#pragma unroll
    for (int ks = 0; ks < 2; ++ks) {
      v16bf bk0 = ldfrag2(&Ks[li][ks * 32 + 8 * g], &Ks[li][ks * 32 + 16 + 8 * g]);
      v16bf bk1 = ldfrag2(&Ks[16 + li][ks * 32 + 8 * g], &Ks[16 + li][ks * 32 + 16 + 8 * g]);
      s0 = wmma_bf16(qf[ks], bk0, s0);
      s1 = wmma_bf16(qf[ks], bk1, s1);
    }

    // online softmax; rows live across the 16 lanes of one half-wave,
    // reductions via DPP16 (pure VALU)
    float corr[8];
#pragma unroll
    for (int r = 0; r < 8; ++r) {
      const int qi = qr0 + r + 8 * g;
      const float a0 = (kv0 + li <= qi)      ? s0[r] : -1.0e9f;
      const float a1 = (kv0 + 16 + li <= qi) ? s1[r] : -1.0e9f;
      const float mx = row_max16(fmaxf(a0, a1));
      const float mnew = fmaxf(mrow[r], mx);
      const float c = __expf(mrow[r] - mnew);
      const float e0 = __expf(a0 - mnew);
      const float e1 = __expf(a1 - mnew);
      const float sum = row_sum16(e0 + e1);
      lrow[r] = lrow[r] * c + sum;
      mrow[r] = mnew;
      corr[r] = c;
      Ps[wave][r + 8 * g][li]      = f2bf(e0);
      Ps[wave][r + 8 * g][16 + li] = f2bf(e1);
    }
#pragma unroll
    for (int dt = 0; dt < 4; ++dt)
#pragma unroll
      for (int r = 0; r < 8; ++r) oacc[dt][r] *= corr[r];

    // P back as A fragment (intra-wave LDS round trip), then O += P*V
    v16bf pf = ldfrag2(&Ps[wave][li][8 * g], &Ps[wave][li][16 + 8 * g]);
#pragma unroll
    for (int dt = 0; dt < 4; ++dt) {
      v16bf bv = ldfrag2(&Vs[dt * 16 + li][8 * g], &Vs[dt * 16 + li][16 + 8 * g]);
      oacc[dt] = wmma_bf16(pf, bv, oacc[dt]);
    }
  }

  // normalize + store o (bf16, [B*L][HID] so the final GEMM consumes it)
  float inv[8];
#pragma unroll
  for (int r = 0; r < 8; ++r) inv[r] = 1.0f / lrow[r];
#pragma unroll
  for (int dt = 0; dt < 4; ++dt)
#pragma unroll
    for (int r = 0; r < 8; ++r) {
      const int row = rowbase + qr0 + r + 8 * g;
      const int col = hh * HD + dt * 16 + li;
      o[(size_t)row * HID + col] = f2bf(oacc[dt][r] * inv[r]);
    }
}

// ---------------------------------------------------------------------------
extern "C" void kernel_launch(void* const* d_in, const int* in_sizes, int n_in,
                              void* d_out, int out_size, void* d_ws, size_t ws_size,
                              hipStream_t stream) {
  (void)in_sizes; (void)n_in; (void)out_size; (void)ws_size;
  const float* x     = (const float*)d_in[0];
  // d_in[1] = mask (causality computed analytically)
  const float* Wq    = (const float*)d_in[2];
  const float* bq    = (const float*)d_in[3];
  const float* Wk    = (const float*)d_in[4];
  const float* bk    = (const float*)d_in[5];
  const float* Wv    = (const float*)d_in[6];
  const float* bv    = (const float*)d_in[7];
  const float* Wo    = (const float*)d_in[8];
  const float* bo    = (const float*)d_in[9];
  const float* gamma = (const float*)d_in[10];
  const float* beta  = (const float*)d_in[11];
  float* out = (float*)d_out;

  const size_t ACT = (size_t)MROWS * HID * sizeof(bf16_t);   // 16 MiB
  const size_t WT  = (size_t)HID * HID * sizeof(bf16_t);     //  2 MiB
  char* ws = (char*)d_ws;
  bf16_t* h   = (bf16_t*)ws;             ws += ACT;
  bf16_t* Wtq = (bf16_t*)ws;             ws += WT;
  bf16_t* Wtk = (bf16_t*)ws;             ws += WT;
  bf16_t* Wtv = (bf16_t*)ws;             ws += WT;
  bf16_t* Wto = (bf16_t*)ws;             ws += WT;
  bf16_t* qb  = (bf16_t*)ws;             ws += ACT;
  bf16_t* kb  = (bf16_t*)ws;             ws += ACT;
  bf16_t* vb  = (bf16_t*)ws;             ws += ACT;
  bf16_t* ob  = (bf16_t*)ws;             ws += ACT;

  ln_kernel<<<MROWS, 256, 0, stream>>>(x, gamma, beta, h);

  dim3 tg(HID / 32, HID / 32), tb(32, 8);
  wtrans_kernel<<<tg, tb, 0, stream>>>(Wq, Wtq);
  wtrans_kernel<<<tg, tb, 0, stream>>>(Wk, Wtk);
  wtrans_kernel<<<tg, tb, 0, stream>>>(Wv, Wtv);
  wtrans_kernel<<<tg, tb, 0, stream>>>(Wo, Wto);

  dim3 gg(HID / 128, MROWS / 128);   // (8, 64)
  gemm128_kernel<false><<<gg, 256, 0, stream>>>(h, Wtq, bq, nullptr, qb, nullptr);
  gemm128_kernel<false><<<gg, 256, 0, stream>>>(h, Wtk, bk, nullptr, kb, nullptr);
  gemm128_kernel<false><<<gg, 256, 0, stream>>>(h, Wtv, bv, nullptr, vb, nullptr);

  flash_kernel<<<dim3(SEQ / 128, BATCH * NH), 256, 0, stream>>>(qb, kb, vb, ob);

  gemm128_kernel<true><<<gg, 256, 0, stream>>>(ob, Wto, bo, x, nullptr, out);
}
